// SelfAttention_1984274890929
// MI455X (gfx1250) — matrix operand order repaired and checked by reading, not compiled, not run on hardware
//
#include <hip/hip_runtime.h>
#include <hip/hip_bf16.h>

// ---------------- CDNA5 WMMA / TDM types ----------------
typedef __attribute__((ext_vector_type(16))) __bf16 v16bf;
typedef __attribute__((ext_vector_type(2)))  __bf16 v2bf;
typedef __attribute__((ext_vector_type(8)))  float  v8f;
typedef unsigned int  u32x4 __attribute__((ext_vector_type(4)));
typedef int           i32x4 __attribute__((ext_vector_type(4)));
typedef int           i32x8 __attribute__((ext_vector_type(8)));

union V16BF {
    v16bf v;
    unsigned int u[8];
};

// ---- f32 -> bf16 conversions: prefer HW packed cvt, else native casts ----
#if __has_builtin(__builtin_amdgcn_cvt_pk_bf16_f32)
__device__ __forceinline__ unsigned int pack2bf(float a, float b) {
    v2bf p = __builtin_amdgcn_cvt_pk_bf16_f32(a, b);
    return __builtin_bit_cast(unsigned int, p);
}
__device__ __forceinline__ unsigned short f2bf(float f) {
    v2bf p = __builtin_amdgcn_cvt_pk_bf16_f32(f, f);
    return (unsigned short)(__builtin_bit_cast(unsigned int, p) & 0xFFFFu);
}
#else
__device__ __forceinline__ unsigned short f2bf(float f) {
    __bf16 h = (__bf16)f;                       // clang lowers to target cvt
    return __builtin_bit_cast(unsigned short, h);
}
__device__ __forceinline__ unsigned int pack2bf(float a, float b) {
    v2bf p = { (__bf16)a, (__bf16)b };
    return __builtin_bit_cast(unsigned int, p);
}
#endif

__device__ __forceinline__ v8f wmma_bf16(v16bf a, v16bf b, v8f c) {
    return __builtin_amdgcn_wmma_f32_16x16x32_bf16(false, a, false, b, (short)0, c, false, false);
}

#define B_DIM 8
#define S_DIM 2048
#define E_DIM 512
#define H_DIM 64
#define NROWS (B_DIM * S_DIM)   // 16384

// ---------------------------------------------------------------------
// Tensor Data Mover: 2-D tile global -> LDS.  Rows are 128B in memory;
// pad_interval = 32 DWORDs (one row) and pad_amount = 4 DWORDs give a
// 144B padded LDS row stride (bank-conflict-free 16-lane row access).
// dsz_code: 0=1B, 1=2B, 2=4B elements.  Dims/strides in elements.
// ---------------------------------------------------------------------
__device__ __forceinline__ void tdm_load_2d(
    const void* gptr, unsigned lds_off, unsigned dsz_code,
    unsigned tensor_d0, unsigned tensor_d1,
    unsigned tile_d0, unsigned tile_d1,
    unsigned long long stride0_elems)
{
    unsigned long long ga = (unsigned long long)(size_t)gptr;
    u32x4 g0;
    g0[0] = 1u;                                             // count=1, user mode
    g0[1] = lds_off;                                        // lds_addr (bytes)
    g0[2] = (unsigned)(ga & 0xFFFFFFFFu);                   // global_addr[31:0]
    g0[3] = (unsigned)((ga >> 32) & 0x01FFFFFFu) | (2u << 30); // addr[56:32] | type=2
    i32x8 g1;
    g1[0] = (int)((dsz_code << 16)  // data_size
                | (1u << 20)        // pad_enable
                | (4u << 22)        // pad_interval: 32 DWORDs (=128B tile row)
                | (3u << 25));      // pad_amount:   4 DWORDs  (=16B pad)
    g1[1] = (int)((tensor_d0 & 0xFFFFu) << 16);             // tensor_dim0[15:0]
    g1[2] = (int)((tensor_d0 >> 16) | ((tensor_d1 & 0xFFFFu) << 16));
    g1[3] = (int)((tensor_d1 >> 16) | ((tile_d0 & 0xFFFFu) << 16));
    g1[4] = (int)(tile_d1 & 0xFFFFu);                       // tile_dim1, tile_dim2=0
    g1[5] = (int)(unsigned)(stride0_elems & 0xFFFFFFFFu);   // tensor_dim0_stride
    g1[6] = (int)(unsigned)((stride0_elems >> 32) & 0xFFFFu);
    g1[7] = 0;
    i32x4 z4 = {0, 0, 0, 0};
#if defined(__clang_major__) && (__clang_major__ >= 23)
    i32x8 z8 = {0, 0, 0, 0, 0, 0, 0, 0};
    __builtin_amdgcn_tensor_load_to_lds(g0, g1, z4, z4, z8, 0);
#else
    __builtin_amdgcn_tensor_load_to_lds(g0, g1, z4, z4, 0);
#endif
}

// =====================================================================
// Kernel 1: Q/K/V projections with double-buffered TDM staging of x.
// x[B*S, E] (f32) @ W[E, H] (f32) -> bf16.  Q, K stored [B*S, H];
// V stored transposed [B][H][S].  1/sqrt(E) folded into Q.
// =====================================================================
__global__ __launch_bounds__(256) void proj_qkv_kernel(
    const float* __restrict__ x,
    const float* __restrict__ Wk,
    const float* __restrict__ Wq,
    const float* __restrict__ Wv,
    unsigned short* __restrict__ Qb,
    unsigned short* __restrict__ Kb,
    unsigned short* __restrict__ Vt)
{
    // W chunk transposed: Wt[mat][h][e], rows padded to 40 halves (80B)
    __shared__ unsigned short Wt[3][64][40];   // 15360 B
    // x slab double buffer: 128 rows x 32 f32, rows padded to 36 (144B)
    __shared__ float xb[2][128][36];           // 36864 B

    const int tid  = threadIdx.x;
    const int wave = tid >> 5;
    const int lane = tid & 31;
    const int ln   = lane & 15;
    const int hg   = lane >> 4;
    const int rblk = blockIdx.x * 128;
    const int r0   = rblk + wave * 16;

    v8f accK[4], accQ[4], accV[4];
    #pragma unroll
    for (int t = 0; t < 4; ++t) { accK[t] = {}; accQ[t] = {}; accV[t] = {}; }

    // prologue: DMA x slab for e0=0 into buffer 0
    if (wave == 0) {
        tdm_load_2d(x + (size_t)rblk * E_DIM, (unsigned)(size_t)&xb[0][0][0],
                    /*f32*/2, 32, 128, 32, 128, E_DIM);
    }

    int cur = 0;
    for (int e0 = 0; e0 < E_DIM; e0 += 32) {
        __syncthreads();   // Wt and xb[cur^1] readers from prev iter done
        // stage 3 x (32 x 64) W chunk transposed, f32 -> bf16
        for (int i = tid; i < 3 * 32 * 64; i += 256) {
            int mat = i >> 11;
            int rem = i & 2047;
            int k   = rem >> 6;       // e within chunk
            int n   = rem & 63;       // h
            const float* src = (mat == 0) ? Wk : ((mat == 1) ? Wq : Wv);
            Wt[mat][n][k] = f2bf(src[(size_t)(e0 + k) * H_DIM + n]);
        }
        if (wave == 0) {
            if (e0 + 32 < E_DIM) {
                tdm_load_2d(x + (size_t)rblk * E_DIM + (e0 + 32),
                            (unsigned)(size_t)&xb[cur ^ 1][0][0],
                            /*f32*/2, 32, 128, 32, 128, E_DIM);
                __builtin_amdgcn_s_wait_tensorcnt(1);   // current slab resident
            } else {
                __builtin_amdgcn_s_wait_tensorcnt(0);
            }
        }
        __syncthreads();

        // A tile from LDS x slab: contiguous 8-float runs
        V16BF a;
        {
            const float* px = &xb[cur][wave * 16 + ln][hg * 16];
            #pragma unroll
            for (int j = 0; j < 4; ++j) a.u[j]     = pack2bf(px[2*j],      px[2*j + 1]);
            #pragma unroll
            for (int j = 0; j < 4; ++j) a.u[4 + j] = pack2bf(px[8 + 2*j], px[8 + 2*j + 1]);
        }

        // Per matrix: batch-load 4 B tiles, then 4 WMMAs
        #pragma unroll
        for (int mat = 0; mat < 3; ++mat) {
            V16BF bt[4];
            #pragma unroll
            for (int t = 0; t < 4; ++t) {
                const unsigned int* p =
                    (const unsigned int*)&Wt[mat][t * 16 + ln][hg * 16];
                #pragma unroll
                for (int j = 0; j < 8; ++j) bt[t].u[j] = p[j];
            }
            v8f* acc = (mat == 0) ? accK : ((mat == 1) ? accQ : accV);
            #pragma unroll
            for (int t = 0; t < 4; ++t)
                acc[t] = wmma_bf16(a.v, bt[t].v, acc[t]);
        }
        cur ^= 1;
    }

    // C layout: row m = v + 8*hg, col n = t*16 + ln
    const float qscale = 0.04419417382415922f;   // 1/sqrt(512)
    const int bb = r0 >> 11;       // batch (blocks never straddle batches)
    const int s0 = r0 & 2047;      // seq position of row 0
    #pragma unroll
    for (int t = 0; t < 4; ++t) {
        #pragma unroll
        for (int v = 0; v < 8; ++v) {
            int m = v + 8 * hg;
            size_t idx = (size_t)(r0 + m) * H_DIM + t * 16 + ln;
            Kb[idx] = f2bf(accK[t][v]);
            Qb[idx] = f2bf(accQ[t][v] * qscale);
        }
        // V transposed: one 16B store of 8 consecutive s for fixed h
        uint4 vp;
        vp.x = pack2bf(accV[t][0], accV[t][1]);
        vp.y = pack2bf(accV[t][2], accV[t][3]);
        vp.z = pack2bf(accV[t][4], accV[t][5]);
        vp.w = pack2bf(accV[t][6], accV[t][7]);
        unsigned short* dst =
            Vt + ((size_t)bb * H_DIM + t * 16 + ln) * S_DIM + s0 + 8 * hg;
        *(uint4*)dst = vp;
    }
}

// =====================================================================
// Kernel 2: flash attention with double-buffered TDM staging of K/V.
// Chunk i+1's tiles DMA while chunk i computes; s_wait_tensorcnt 2
// tolerates the 2 in-flight descriptors (TDM completes in order).
// =====================================================================
__global__ __launch_bounds__(256) void flash_attn_kernel(
    const unsigned short* __restrict__ Qb,
    const unsigned short* __restrict__ Kb,
    const unsigned short* __restrict__ Vt,
    float* __restrict__ out)
{
    __shared__ unsigned short Kc[2][64][72];    // [buf][key][h]  18432 B
    __shared__ unsigned short Vc[2][64][72];    // [buf][h][key]  18432 B
    __shared__ unsigned short Pb[8][16][72];    // per-wave P staging 18432 B

    const int tid  = threadIdx.x;
    const int wave = tid >> 5;
    const int lane = tid & 31;
    const int ln   = lane & 15;
    const int hg   = lane >> 4;
    const int b    = blockIdx.x >> 4;
    const int q0   = (blockIdx.x & 15) * 128 + wave * 16;
    const size_t rowQ = (size_t)b * S_DIM + q0;

    const unsigned short* Kbase = Kb + (size_t)b * S_DIM * H_DIM;
    const unsigned short* Vbase = Vt + (size_t)b * H_DIM * S_DIM;

    // ---- Q in A-layout registers (2 tiles over H=64) ----
    V16BF qa[2];
    #pragma unroll
    for (int c = 0; c < 2; ++c) {
        const unsigned int* pq =
            (const unsigned int*)(Qb + (rowQ + ln) * H_DIM + c * 32 + hg * 16);
        #pragma unroll
        for (int j = 0; j < 4; ++j) qa[c].u[j]     = pq[j];
        #pragma unroll
        for (int j = 0; j < 4; ++j) qa[c].u[4 + j] = pq[4 + j];
    }

    float M[8], L[8];
    #pragma unroll
    for (int v = 0; v < 8; ++v) { M[v] = -__builtin_inff(); L[v] = 0.0f; }
    v8f O[4];
    #pragma unroll
    for (int t = 0; t < 4; ++t) O[t] = {};

    const float log2e = 1.4426950408889634f;

    // prologue: DMA chunk 0 into buffer 0
    if (wave == 0) {
        tdm_load_2d(Kbase, (unsigned)(size_t)&Kc[0][0][0],
                    /*bf16*/1, H_DIM, S_DIM, H_DIM, 64, H_DIM);
        tdm_load_2d(Vbase, (unsigned)(size_t)&Vc[0][0][0],
                    /*bf16*/1, 64, H_DIM, 64, H_DIM, S_DIM);
    }

    int cur = 0;
    for (int k0 = 0; k0 < S_DIM; k0 += 64) {
        __syncthreads();   // previous compute done with buf[cur^1]
        if (wave == 0) {
            if (k0 + 64 < S_DIM) {
                tdm_load_2d(Kbase + (size_t)(k0 + 64) * H_DIM,
                            (unsigned)(size_t)&Kc[cur ^ 1][0][0],
                            1, H_DIM, S_DIM, H_DIM, 64, H_DIM);
                tdm_load_2d(Vbase + (k0 + 64),
                            (unsigned)(size_t)&Vc[cur ^ 1][0][0],
                            1, 64, H_DIM, 64, H_DIM, S_DIM);
                __builtin_amdgcn_s_wait_tensorcnt(2);   // current chunk resident
            } else {
                __builtin_amdgcn_s_wait_tensorcnt(0);
            }
        }
        __syncthreads();

        // ---- S = Q K^T (scale folded into Q): batch loads then WMMAs ----
        V16BF kbt[4][2];
        #pragma unroll
        for (int t = 0; t < 4; ++t) {
            #pragma unroll
            for (int c = 0; c < 2; ++c) {
                const unsigned int* pk =
                    (const unsigned int*)&Kc[cur][t * 16 + ln][c * 32 + hg * 16];
                #pragma unroll
                for (int j = 0; j < 8; ++j) kbt[t][c].u[j] = pk[j];
            }
        }
        v8f Sc[4];
        #pragma unroll
        for (int t = 0; t < 4; ++t) {
            v8f acc = {};
            acc = wmma_bf16(qa[0].v, kbt[t][0].v, acc);
            acc = wmma_bf16(qa[1].v, kbt[t][1].v, acc);
            Sc[t] = acc;
        }

        // ---- online softmax (row = v + 8*hg, cols across 16-lane group) ----
        float alpha[8];
        #pragma unroll
        for (int v = 0; v < 8; ++v) {
            float mx = fmaxf(fmaxf(Sc[0][v], Sc[1][v]), fmaxf(Sc[2][v], Sc[3][v]));
            #pragma unroll
            for (int off = 8; off >= 1; off >>= 1)
                mx = fmaxf(mx, __shfl_xor(mx, off, 16));
            float Mn = fmaxf(M[v], mx);
            float a  = exp2f((M[v] - Mn) * log2e);
            float rs = 0.0f;
            #pragma unroll
            for (int t = 0; t < 4; ++t) {
                float p = exp2f((Sc[t][v] - Mn) * log2e);
                Sc[t][v] = p;
                rs += p;
            }
            #pragma unroll
            for (int off = 8; off >= 1; off >>= 1)
                rs += __shfl_xor(rs, off, 16);
            L[v] = L[v] * a + rs;
            M[v] = Mn;
            alpha[v] = a;
        }
        #pragma unroll
        for (int t = 0; t < 4; ++t) {
            #pragma unroll
            for (int v = 0; v < 8; ++v) O[t][v] *= alpha[v];
        }

        // ---- P: C-format f32 -> A-format bf16 via per-wave LDS ----
        #pragma unroll
        for (int t = 0; t < 4; ++t) {
            #pragma unroll
            for (int v = 0; v < 8; ++v)
                Pb[wave][v + 8 * hg][t * 16 + ln] = f2bf(Sc[t][v]);
        }
        __syncthreads();

        V16BF pa[2];
        #pragma unroll
        for (int c = 0; c < 2; ++c) {
            const unsigned int* pp =
                (const unsigned int*)&Pb[wave][ln][c * 32 + hg * 16];
            #pragma unroll
            for (int j = 0; j < 4; ++j) pa[c].u[j]     = pp[j];
            #pragma unroll
            for (int j = 0; j < 4; ++j) pa[c].u[4 + j] = pp[4 + j];
        }

        // ---- O += P V : batch B-tile loads from Vc[h][key], then WMMAs ----
        V16BF vbt[4][2];
        #pragma unroll
        for (int ht = 0; ht < 4; ++ht) {
            #pragma unroll
            for (int c = 0; c < 2; ++c) {
                const unsigned int* pv =
                    (const unsigned int*)&Vc[cur][ht * 16 + ln][c * 32 + hg * 16];
                #pragma unroll
                for (int j = 0; j < 8; ++j) vbt[ht][c].u[j] = pv[j];
            }
        }
        #pragma unroll
        for (int ht = 0; ht < 4; ++ht) {
            O[ht] = wmma_bf16(pa[0].v, vbt[ht][0].v, O[ht]);
            O[ht] = wmma_bf16(pa[1].v, vbt[ht][1].v, O[ht]);
        }

        cur ^= 1;
    }

    // ---- normalize and store f32 output ----
    #pragma unroll
    for (int v = 0; v < 8; ++v) {
        float inv = 1.0f / L[v];
        int m = v + 8 * hg;
        #pragma unroll
        for (int ht = 0; ht < 4; ++ht)
            out[(rowQ + m) * H_DIM + ht * 16 + ln] = O[ht][v] * inv;
    }
}

// =====================================================================
extern "C" void kernel_launch(void* const* d_in, const int* in_sizes, int n_in,
                              void* d_out, int out_size, void* d_ws, size_t ws_size,
                              hipStream_t stream) {
    (void)in_sizes; (void)n_in; (void)out_size; (void)ws_size;
    // setup_inputs order: x, Wk, Wq, Wv
    const float* x  = (const float*)d_in[0];
    const float* Wk = (const float*)d_in[1];
    const float* Wq = (const float*)d_in[2];
    const float* Wv = (const float*)d_in[3];

    unsigned short* Qb = (unsigned short*)d_ws;                    // 2 MB
    unsigned short* Kb = Qb + (size_t)NROWS * H_DIM;               // 2 MB
    unsigned short* Vt = Kb + (size_t)NROWS * H_DIM;               // 2 MB, [B][H][S]

    proj_qkv_kernel<<<NROWS / 128, 256, 0, stream>>>(x, Wk, Wq, Wv, Qb, Kb, Vt);
    flash_attn_kernel<<<NROWS / 128, 256, 0, stream>>>(Qb, Kb, Vt, (float*)d_out);
}
